// ProteinFeaturesLigand_5781025980979
// MI455X (gfx1250) — matrix operand order repaired and checked by reading, not compiled, not run on hardware
//
#include <hip/hip_runtime.h>

#define Lq 1024
#define Bq 2
#define Mq 25

typedef __attribute__((ext_vector_type(16))) __bf16 v16bf;
typedef __attribute__((ext_vector_type(8)))  __bf16 v8bf;
typedef __attribute__((ext_vector_type(8)))  float  v8f;

__constant__ int c_PA[24] = {0,2,3,4,1,1,1,1,0,0,0,4,4,3,0,2,3,4,2,3,4,2,3,2};
__constant__ int c_PB[24] = {0,2,3,4,0,2,3,4,2,3,4,2,3,2,1,1,1,1,0,0,0,4,4,3};

__device__ __forceinline__ float rbff(float d, int r){
    float mu = 2.0f + (20.0f/15.0f)*(float)r;
    float z  = (d - mu) * 0.8f;            // sigma = 1.25
    return __expf(-z*z);
}
__device__ __forceinline__ float wred16(float v){
    v += __shfl_xor(v, 1, 32); v += __shfl_xor(v, 2, 32);
    v += __shfl_xor(v, 4, 32); v += __shfl_xor(v, 8, 32);
    return v;
}
__device__ __forceinline__ float wred32(float v){
    v = wred16(v); v += __shfl_xor(v, 16, 32); return v;
}
// A-fragment load for 16-bit 16x32 A: lane holds 16 contiguous elems (two 8-elem chunks).
__device__ __forceinline__ v16bf ldA16(const __bf16* p){
    v8bf lo = *(const v8bf*)(p);
    v8bf hi = *(const v8bf*)(p + 8);
    return __builtin_shufflevector(lo, hi, 0,1,2,3,4,5,6,7,8,9,10,11,12,13,14,15);
}
// CDNA5 async copy: 16B global -> LDS per lane, tracked by ASYNCcnt.
__device__ __forceinline__ void async_g2l_b128(unsigned lds_addr, const void* gaddr){
    asm volatile("global_load_async_to_lds_b128 %0, %1, off"
                 :: "v"(lds_addr), "v"(gaddr) : "memory");
}
__device__ __forceinline__ unsigned lds_off(const void* p){
    return (unsigned)(size_t)p;   // LDS aperture keeps the byte offset in addr[31:0]
}

// Fused layernorm epilogue over a 16x128 row block held in 8 f32 WMMA accumulators.
// Row M = r + 8*(lane>=16) lives entirely in one 16-lane half -> half-wave reductions.
__device__ __forceinline__ void ln_store128(v8f acc[8], const float* bias,
                                            const float* gam, const float* bet,
                                            float* out, int rowBase, int lane){
    int ncol = lane & 15;
    int hf   = lane >> 4;
    float gv[8], bv[8];
#pragma unroll
    for (int n = 0; n < 8; n++){ gv[n] = gam[n*16+ncol]; bv[n] = bet[n*16+ncol]; }
    if (bias){
#pragma unroll
        for (int n = 0; n < 8; n++){
            float bb = bias[n*16+ncol];
#pragma unroll
            for (int r = 0; r < 8; r++) acc[n][r] += bb;
        }
    }
#pragma unroll
    for (int r = 0; r < 8; r++){
        float s = 0.f;
#pragma unroll
        for (int n = 0; n < 8; n++) s += acc[n][r];
        s = wred16(s);
        float mean = s * (1.0f/128.0f);
        float q = 0.f;
#pragma unroll
        for (int n = 0; n < 8; n++){ float d = acc[n][r] - mean; q += d*d; }
        q = wred16(q);
        float rstd = rsqrtf(q * (1.0f/128.0f) + 1e-5f);
        float* orow = out + (size_t)(rowBase + r + hf*8) * 128;
#pragma unroll
        for (int n = 0; n < 8; n++)
            orow[n*16+ncol] = (acc[n][r] - mean) * rstd * gv[n] + bv[n];
    }
}

// ---------------- K0: transpose weights to bf16 [N=128][K] for B fragments --------
__global__ void __launch_bounds__(256)
prep_kernel(const float* __restrict__ edgeW, const float* __restrict__ npdW,
            const float* __restrict__ yeW,
            __bf16* __restrict__ WTe, __bf16* __restrict__ WTv, __bf16* __restrict__ WTy){
    int tid = blockIdx.x * 256 + threadIdx.x;
    if (tid < 128*416){ int n = tid/416, k = tid%416; WTe[tid] = (__bf16)edgeW[k*128+n]; }
    if (tid < 128*160){ int n = tid/160, k = tid%160; WTv[tid] = (__bf16)(k < 148 ? npdW[k*128+n] : 0.0f); }
    if (tid < 128*32 ){ int n = tid/32 , k = tid%32 ; WTy[tid] = (__bf16)(k < 16  ? yeW[k*128+n] : 0.0f); }
}

// ---------------- K1: per-residue Cb + local frame (12 floats) --------------------
__global__ void __launch_bounds__(256)
frames_kernel(const float* __restrict__ X, float* __restrict__ fr){
    int idx = blockIdx.x * 256 + threadIdx.x;
    if (idx >= Bq*Lq) return;
    const float* xp = X + (size_t)idx * 12;
    float Nx = xp[0], Ny = xp[1], Nz = xp[2];
    float Ax = xp[3], Ay = xp[4], Az = xp[5];   // Ca
    float Cx = xp[6], Cy = xp[7], Cz = xp[8];
    float bx = Ax-Nx, by = Ay-Ny, bz = Az-Nz;
    float cx = Cx-Ax, cy = Cy-Ay, cz = Cz-Az;
    float ax = by*cz - bz*cy, ay = bz*cx - bx*cz, az = bx*cy - by*cx;
    float Cbx = -0.58273431f*ax + 0.56802827f*bx - 0.54067466f*cx + Ax;
    float Cby = -0.58273431f*ay + 0.56802827f*by - 0.54067466f*cy + Ay;
    float Cbz = -0.58273431f*az + 0.56802827f*bz - 0.54067466f*cz + Az;
    float ux = -bx, uy = -by, uz = -bz;
    float n1 = fmaxf(sqrtf(ux*ux+uy*uy+uz*uz), 1e-12f);
    float e1x = ux/n1, e1y = uy/n1, e1z = uz/n1;
    float pr = e1x*cx + e1y*cy + e1z*cz;
    float wx = cx - pr*e1x, wy = cy - pr*e1y, wz = cz - pr*e1z;
    float n2 = fmaxf(sqrtf(wx*wx+wy*wy+wz*wz), 1e-12f);
    float e2x = wx/n2, e2y = wy/n2, e2z = wz/n2;
    float e3x = e1y*e2z - e1z*e2y, e3y = e1z*e2x - e1x*e2z, e3z = e1x*e2y - e1y*e2x;
    float* o = fr + (size_t)idx * 12;
    o[0]=Cbx; o[1]=Cby; o[2]=Cbz;
    o[3]=e1x; o[4]=e1y; o[5]=e1z;
    o[6]=e2x; o[7]=e2y; o[8]=e2z;
    o[9]=e3x; o[10]=e3y; o[11]=e3z;
}

// ---------------- K2: masked distance row + top-32 (stable, lower-index ties) -----
__global__ void __launch_bounds__(256)
topk_kernel(const float* __restrict__ X, const float* __restrict__ mask,
            int* __restrict__ Eidx, float* __restrict__ Dnb, float* __restrict__ EidxF){
    __shared__ float sD[Lq];
    __shared__ float sM[Lq];
    __shared__ float rv[256];
    __shared__ int   ri[256];
    int b = blockIdx.y, i = blockIdx.x, tid = threadIdx.x;
    const float* Ci = X + ((size_t)(b*Lq + i)*4 + 1)*3;
    float cx = Ci[0], cy = Ci[1], cz = Ci[2];
    float mi = mask[b*Lq + i];
    float lmax = -1e30f;
    for (int j = tid; j < Lq; j += 256){
        const float* Cj = X + ((size_t)(b*Lq + j)*4 + 1)*3;
        float dx = cx-Cj[0], dy = cy-Cj[1], dz = cz-Cj[2];
        float m2 = mi * mask[b*Lq + j];
        float d  = m2 * sqrtf(dx*dx + dy*dy + dz*dz + 1e-6f);
        sD[j] = d; sM[j] = m2;
        lmax = fmaxf(lmax, d);
    }
    rv[tid] = lmax; __syncthreads();
    for (int s = 128; s > 0; s >>= 1){
        if (tid < s) rv[tid] = fmaxf(rv[tid], rv[tid+s]);
        __syncthreads();
    }
    float Dmax = rv[0];
    __syncthreads();
    for (int j = tid; j < Lq; j += 256) sD[j] += (1.0f - sM[j]) * Dmax;
    __syncthreads();
    for (int t = 0; t < 32; t++){
        float bvv = 3.4e38f; int bii = 0x7fffffff;
        for (int j = tid; j < Lq; j += 256){
            float v = sD[j];
            if (v < bvv || (v == bvv && j < bii)){ bvv = v; bii = j; }
        }
        rv[tid] = bvv; ri[tid] = bii; __syncthreads();
        for (int s = 128; s > 0; s >>= 1){
            if (tid < s){
                float v2 = rv[tid+s]; int i2 = ri[tid+s];
                if (v2 < rv[tid] || (v2 == rv[tid] && i2 < ri[tid])){ rv[tid]=v2; ri[tid]=i2; }
            }
            __syncthreads();
        }
        if (tid == 0){
            int jm = ri[0];
            size_t o = (size_t)(b*Lq + i)*32 + t;
            Eidx[o] = jm; Dnb[o] = rv[0]; EidxF[o] = (float)jm;
            sD[jm] = 3.0e38f;
        }
        __syncthreads();
    }
}

// ---------------- K3: edge feature rows (416 bf16 per edge) -----------------------
__global__ void __launch_bounds__(256)
edge_feat_kernel(const float* __restrict__ X, const float* __restrict__ fr,
                 const int* __restrict__ Eidx, const float* __restrict__ Dnb,
                 const int* __restrict__ Ridx, const int* __restrict__ chains,
                 const float* __restrict__ posW, const float* __restrict__ posb,
                 __bf16* __restrict__ Fe){
    __shared__ float sat[8][2][15];
    __shared__ float sdist[8][25];
    int lane = threadIdx.x & 31, w = threadIdx.x >> 5;
    int row = blockIdx.x * 8 + w;          // < 65536
    int bi = row >> 5;
    int i = bi & (Lq-1), b = bi >> 10;
    int j = Eidx[row];
    if (lane < 30){
        int which = lane / 15, c = lane % 15;
        int res = which ? j : i;
        int a = c / 3, comp = c % 3;
        float v = (a < 4) ? X[((size_t)(b*Lq + res)*4 + a)*3 + comp]
                          : fr[(size_t)(b*Lq + res)*12 + comp];   // Cb
        sat[w][which][c] = v;
    }
    __syncthreads();
    if (lane == 0) sdist[w][0] = Dnb[row];
    else if (lane < 25){
        int p = lane - 1, aa = c_PA[p], bb = c_PB[p];
        float dx = sat[w][0][aa*3+0] - sat[w][1][bb*3+0];
        float dy = sat[w][0][aa*3+1] - sat[w][1][bb*3+1];
        float dz = sat[w][0][aa*3+2] - sat[w][1][bb*3+2];
        sdist[w][lane] = sqrtf(dx*dx + dy*dy + dz*dz + 1e-6f);
    }
    __syncthreads();
    int riv = Ridx[b*Lq + i], rjv = Ridx[b*Lq + j];
    int ec  = (chains[b*Lq + i] == chains[b*Lq + j]) ? 1 : 0;
    int off = riv - rjv;
    int dd  = ec ? min(max(off + 32, 0), 64) : 65;
    __bf16* Frow = Fe + (size_t)row * 416;
    for (int f = lane; f < 416; f += 32){
        float v;
        if (f < 16) v = posW[dd*16 + f] + posb[f];
        else { int p = (f-16) >> 4, r = (f-16) & 15; v = rbff(sdist[w][p], r); }
        Frow[f] = (__bf16)v;
    }
}

// ---------------- K4: node feature rows (160 bf16 per (b,l,m), K padded) ----------
__global__ void __launch_bounds__(256)
node_feat_kernel(const float* __restrict__ X, const float* __restrict__ fr,
                 const float* __restrict__ Y, const int* __restrict__ Yt,
                 const int* __restrict__ ptab, const float* __restrict__ typeW,
                 const float* __restrict__ typeb, __bf16* __restrict__ Fv){
    int lane = threadIdx.x & 31, w = threadIdx.x >> 5;
    int row = blockIdx.x * 8 + w;          // < 51200
    int q = row / Mq;
    int l = q & (Lq-1), b = q >> 10;
    const float* yp = Y + (size_t)row * 3;
    float yx = yp[0], yy = yp[1], yz = yp[2];
    size_t rr = (size_t)(b*Lq + l);
    float ax[5], ay[5], az[5];
#pragma unroll
    for (int a = 0; a < 4; a++){
        ax[a] = X[(rr*4 + a)*3 + 0];
        ay[a] = X[(rr*4 + a)*3 + 1];
        az[a] = X[(rr*4 + a)*3 + 2];
    }
    const float* f0 = fr + rr*12;
    ax[4] = f0[0]; ay[4] = f0[1]; az[4] = f0[2];
    int t  = Yt[row];
    int p1 = ptab[120 + t];
    int p2 = ptab[240 + t];
    float dx = yx-ax[1], dy = yy-ay[1], dz = yz-az[1];
    float lx = f0[3]*dx + f0[4]*dy + f0[5]*dz;
    float ly = f0[6]*dx + f0[7]*dy + f0[8]*dz;
    float lz = f0[9]*dx + f0[10]*dy + f0[11]*dz;
    float rxy  = sqrtf(lx*lx + ly*ly + 1e-8f);
    float rxyz = sqrtf(lx*lx + ly*ly + lz*lz) + 1e-8f;
    float fang0 = lx/rxy, fang1 = ly/rxy, fang2 = rxy/rxyz, fang3 = lz/rxyz;
    __bf16* Frow = Fv + (size_t)row * 160;
    for (int f = lane; f < 160; f += 32){
        float v;
        if (f < 80){
            int a = f >> 4, r = f & 15;
            float ddx = ax[a]-yx, ddy = ay[a]-yy, ddz = az[a]-yz;
            v = rbff(sqrtf(ddx*ddx + ddy*ddy + ddz*ddz + 1e-6f), r);
        } else if (f < 144){
            int c = f - 80;
            v = typeW[t*64 + c] + typeW[(120+p1)*64 + c] + typeW[(139+p2)*64 + c] + typeb[c];
        } else if (f < 148){
            v = (f == 144) ? fang0 : (f == 145) ? fang1 : (f == 146) ? fang2 : fang3;
        } else v = 0.0f;
        Frow[f] = (__bf16)v;
    }
}

// ---------------- K5: bf16 WMMA GEMM (Nx128) + fused layernorm --------------------
// 32 rows per wave (two 16-row blocks share each B fragment); A tiles are staged
// into LDS with double-buffered global_load_async_to_lds_b128 (ASYNCcnt).
// LDS row pitch = 40 elems (80B): gcd(80,256)=16 -> conflict-free b128 reads.
__global__ void __launch_bounds__(256)
gemm_ln_kernel(const __bf16* __restrict__ A, const __bf16* __restrict__ WT,
               const float* bias, const float* __restrict__ gam,
               const float* __restrict__ bet, float* __restrict__ out, int K){
    __shared__ __align__(16) __bf16 Atile[8][2][32][40];
    int lane = threadIdx.x & 31, wave = threadIdx.x >> 5;
    int rowBase = (blockIdx.x * 8 + wave) * 32;
    v8f acc0[8], acc1[8];
#pragma unroll
    for (int n = 0; n < 8; n++)
#pragma unroll
        for (int r = 0; r < 8; r++){ acc0[n][r] = 0.0f; acc1[n][r] = 0.0f; }
    int ncol = lane & 15;
    int bOff = (lane < 16) ? 0 : 16;
    int cA0  = (lane < 16) ? 0 : 16;
    int T = K >> 5;
    // per-lane async copy mapping: 4 instrs move 32 rows x 64B; lane covers a 16B chunk
    int arl = lane >> 2;                 // row 0..7 (+8 per t)
    int acl = (lane & 3) * 8;            // element column of 16B chunk
    // prologue: tile 0 -> buf 0
#pragma unroll
    for (int t = 0; t < 4; t++){
        int r = arl + t*8;
        async_g2l_b128(lds_off(&Atile[wave][0][r][acl]),
                       A + (size_t)(rowBase + r)*K + acl);
    }
    for (int kt = 0; kt < T; kt++){
        int buf = kt & 1;
        if (kt + 1 < T){
            int kb2 = (kt + 1) << 5;
#pragma unroll
            for (int t = 0; t < 4; t++){
                int r = arl + t*8;
                async_g2l_b128(lds_off(&Atile[wave][buf ^ 1][r][acl]),
                               A + (size_t)(rowBase + r)*K + kb2 + acl);
            }
            asm volatile("s_wait_asynccnt 4" ::: "memory");   // tile kt landed
        } else {
            asm volatile("s_wait_asynccnt 0" ::: "memory");   // last tile landed
        }
        v16bf a0 = ldA16(&Atile[wave][buf][     (lane & 15)][cA0]);
        v16bf a1 = ldA16(&Atile[wave][buf][16 + (lane & 15)][cA0]);
        int kb = kt << 5;
#pragma unroll
        for (int n = 0; n < 8; n++){
            v16bf bb = *(const v16bf*)(WT + (size_t)(n*16 + ncol)*K + kb + bOff);
            acc0[n] = __builtin_amdgcn_wmma_f32_16x16x32_bf16(
                false, a0, false, bb, (short)0, acc0[n], false, false);
            acc1[n] = __builtin_amdgcn_wmma_f32_16x16x32_bf16(
                false, a1, false, bb, (short)0, acc1[n], false, false);
        }
    }
    ln_store128(acc0, bias, gam, bet, out, rowBase,      lane);
    ln_store128(acc1, bias, gam, bet, out, rowBase + 16, lane);
}

// ---------------- K6: fused Y_edges RBF -> WMMA (K=32 padded) -> layernorm --------
__global__ void __launch_bounds__(256)
yedge_kernel(const float* __restrict__ Y, const __bf16* __restrict__ WTy,
             const float* __restrict__ gam, const float* __restrict__ bet,
             float* __restrict__ out){
    __shared__ __align__(64) __bf16 atile[8][16][32];
    __shared__ float sd[8][16];
    int lane = threadIdx.x & 31, wave = threadIdx.x >> 5;
    int g0 = (blockIdx.x * 8 + wave) * 16;        // row base, rows = B*L*M*M
    if (lane < 16){
        int g  = g0 + lane;
        int m2 = g % Mq;  int q  = g / Mq;
        int m  = q % Mq;  int q2 = q / Mq;
        int l  = q2 & (Lq-1), b = q2 >> 10;
        const float* ya = Y + (size_t)(((b*Lq + l)*Mq) + m ) * 3;
        const float* yb = Y + (size_t)(((b*Lq + l)*Mq) + m2) * 3;
        float dx = ya[0]-yb[0], dy = ya[1]-yb[1], dz = ya[2]-yb[2];
        sd[wave][lane] = sqrtf(dx*dx + dy*dy + dz*dz + 1e-6f);
    }
    __syncthreads();
#pragma unroll
    for (int t = 0; t < 16; t++){
        float v = (lane < 16) ? rbff(sd[wave][t], lane) : 0.0f;
        atile[wave][t][lane] = (__bf16)v;
    }
    __syncthreads();
    v8f acc[8];
#pragma unroll
    for (int n = 0; n < 8; n++)
#pragma unroll
        for (int r = 0; r < 8; r++) acc[n][r] = 0.0f;
    int ncol = lane & 15;
    int bOff = (lane < 16) ? 0 : 16;
    v16bf a = ldA16(&atile[wave][lane & 15][(lane < 16) ? 0 : 16]);
#pragma unroll
    for (int n = 0; n < 8; n++){
        v16bf bb = *(const v16bf*)(WTy + (size_t)(n*16 + ncol)*32 + bOff);
        acc[n] = __builtin_amdgcn_wmma_f32_16x16x32_bf16(
            false, a, false, bb, (short)0, acc[n], false, false);
    }
    ln_store128(acc, nullptr, gam, bet, out, g0, lane);
}

// ---------------- K7: Y_nodes (one-hot gather-sum) + layernorm, + Y_m copy --------
__global__ void __launch_bounds__(256)
ynodes_kernel(const int* __restrict__ Yt, const int* __restrict__ ptab,
              const float* __restrict__ ynW, const float* __restrict__ gam,
              const float* __restrict__ bet, const float* __restrict__ Ym,
              float* __restrict__ outN, float* __restrict__ outM){
    int lane = threadIdx.x & 31, w = threadIdx.x >> 5;
    int row = blockIdx.x * 8 + w;   // < 51200
    int t  = Yt[row];
    int p1 = ptab[120 + t];
    int p2 = ptab[240 + t];
    const float* r0 = ynW + (size_t)t*128        + lane*4;
    const float* r1 = ynW + (size_t)(120+p1)*128 + lane*4;
    const float* r2 = ynW + (size_t)(139+p2)*128 + lane*4;
    float v[4];
#pragma unroll
    for (int c = 0; c < 4; c++) v[c] = r0[c] + r1[c] + r2[c];
    float s = v[0]+v[1]+v[2]+v[3];
    s = wred32(s);
    float mean = s * (1.0f/128.0f);
    float q = 0.f;
#pragma unroll
    for (int c = 0; c < 4; c++){ float d = v[c]-mean; q += d*d; }
    q = wred32(q);
    float rstd = rsqrtf(q * (1.0f/128.0f) + 1e-5f);
    float* o = outN + (size_t)row*128 + lane*4;
#pragma unroll
    for (int c = 0; c < 4; c++)
        o[c] = (v[c]-mean)*rstd*gam[lane*4+c] + bet[lane*4+c];
    if (lane == 0) outM[row] = Ym[row];
}

extern "C" void kernel_launch(void* const* d_in, const int* in_sizes, int n_in,
                              void* d_out, int out_size, void* d_ws, size_t ws_size,
                              hipStream_t stream) {
    const float* X      = (const float*)d_in[0];
    const float* Y      = (const float*)d_in[1];
    const float* Ym     = (const float*)d_in[2];
    const int*   Yt     = (const int*)  d_in[3];
    const float* mask   = (const float*)d_in[4];
    const int*   Ridx   = (const int*)  d_in[5];
    const int*   chains = (const int*)  d_in[6];
    const float* posW   = (const float*)d_in[7];
    const float* posb   = (const float*)d_in[8];
    const float* edgeW  = (const float*)d_in[9];
    const float* lneg   = (const float*)d_in[10];
    const float* lneb   = (const float*)d_in[11];
    const float* npdW   = (const float*)d_in[12];
    const float* npdb   = (const float*)d_in[13];
    const float* lnng   = (const float*)d_in[14];
    const float* lnnb   = (const float*)d_in[15];
    const float* typeW  = (const float*)d_in[16];
    const float* typeb  = (const float*)d_in[17];
    const float* ynW    = (const float*)d_in[18];
    const float* yeW    = (const float*)d_in[19];
    const float* lnyeg  = (const float*)d_in[20];
    const float* lnyeb  = (const float*)d_in[21];
    const float* lnyng  = (const float*)d_in[22];
    const float* lnynb  = (const float*)d_in[23];
    const int*   ptab   = (const int*)  d_in[24];

    // Output layout (floats), tuple order (V, E, E_idx, Y_nodes, Y_edges, Y_m)
    float* out     = (float*)d_out;
    float* outV    = out;                // 2*1024*25*128
    float* outE    = out + 6553600;      // 2*1024*32*128
    float* outEidx = out + 14942208;     // 2*1024*32
    float* outYN   = out + 15007744;     // 2*1024*25*128
    float* outYE   = out + 21561344;     // 2*1024*25*25*128
    float* outYM   = out + 185401344;    // 2*1024*25

    // Workspace layout (bytes, all 256B-aligned)
    char* ws = (char*)d_ws;
    float*  frames = (float*) (ws + 0);          //  98,304
    int*    eidx   = (int*)   (ws + 98304);      // 262,144
    float*  dnb    = (float*) (ws + 360448);     // 262,144
    __bf16* WTe    = (__bf16*)(ws + 622592);     // 106,496
    __bf16* WTv    = (__bf16*)(ws + 729088);     //  40,960
    __bf16* WTy    = (__bf16*)(ws + 770048);     //   8,192
    __bf16* Fe     = (__bf16*)(ws + 778240);     // 54,525,952
    __bf16* Fv     = (__bf16*)(ws + 55304192);   // 16,384,000 (end 71,688,192)

    prep_kernel     <<<208, 256, 0, stream>>>(edgeW, npdW, yeW, WTe, WTv, WTy);
    frames_kernel   <<<8, 256, 0, stream>>>(X, frames);
    topk_kernel     <<<dim3(Lq, Bq), 256, 0, stream>>>(X, mask, eidx, dnb, outEidx);
    edge_feat_kernel<<<8192, 256, 0, stream>>>(X, frames, eidx, dnb, Ridx, chains,
                                               posW, posb, Fe);
    node_feat_kernel<<<6400, 256, 0, stream>>>(X, frames, Y, Yt, ptab, typeW, typeb, Fv);
    gemm_ln_kernel  <<<256, 256, 0, stream>>>(Fe, WTe, nullptr, lneg, lneb, outE, 416);
    gemm_ln_kernel  <<<200, 256, 0, stream>>>(Fv, WTv, npdb,    lnng, lnnb, outV, 160);
    yedge_kernel    <<<10000, 256, 0, stream>>>(Y, WTy, lnyeg, lnyeb, outYE);
    ynodes_kernel   <<<6400, 256, 0, stream>>>(Yt, ptab, ynW, lnyng, lnynb, Ym, outYN, outYM);

    (void)in_sizes; (void)n_in; (void)out_size; (void)ws_size;
}
